// PatchLSTMBlock_87840671138302
// MI455X (gfx1250) — matrix operand order repaired and checked by reading, not compiled, not run on hardware
//
#include <hip/hip_runtime.h>

typedef __attribute__((ext_vector_type(16))) _Float16 v16h;
typedef __attribute__((ext_vector_type(8)))  _Float16 v8h;
typedef __attribute__((ext_vector_type(8)))  float    v8f;

#define F_DIM 512
#define H_DIM 1024
#define G_DIM 4096   // 4*H
#define T_DIM 128
#define N_SEQ 256
#define HPAD  (H_DIM + 8)
#define FPAD  (F_DIM + 8)
#define CPAD  (H_DIM + 8)

// ---------------------------------------------------------------- helpers

__device__ __forceinline__ v8f wmma16(v16h a, v16h b, v8f c) {
  // v_wmma_f32_16x16x32_f16: D = A(16x32 f16) x B(32x16 f16) + C(f32)
  return __builtin_amdgcn_wmma_f32_16x16x32_f16(
      /*neg_a=*/false, a, /*neg_b=*/false, b,
      /*c_mod=*/(short)0, c, /*reuse_a=*/false, /*reuse_b=*/false);
}

__device__ __forceinline__ float fast_sigmoid(float x) {
  return 1.0f / (1.0f + __expf(-x));
}
__device__ __forceinline__ float fast_tanh(float x) {
  return 2.0f / (1.0f + __expf(-2.0f * x)) - 1.0f;
}

// A fragment (16x32 f16) from row-major f16 buffer (LDS).
// lanes 0-15: M=lane, K = kbase+0..15
// lanes 16-31: M=lane-16, K = kbase+16..31
__device__ __forceinline__ v16h load_a(const _Float16* base, int lda,
                                       int m, int khalf, int kbase) {
  const _Float16* p = base + m * lda + kbase + khalf * 16;
  v8h lo = *(const v8h*)(p);
  v8h hi = *(const v8h*)(p + 8);
  v16h r;
#pragma unroll
  for (int i = 0; i < 8; ++i) { r[i] = lo[i]; r[i + 8] = hi[i]; }
  return r;
}

// B fragment (32x16 f16) from transposed weight W^T[col][k] (k contiguous).
// lanes 0-15 hold K=kbase+0..15 of column col; lanes 16-31 hold K=kbase+16..31.
// Non-temporal: weight stream has no WGP$ reuse (reuse lives in L2).
__device__ __forceinline__ v16h load_b_nt(const _Float16* colptr, int khalf, int kbase) {
  const v16h* p = (const v16h*)(colptr + kbase + khalf * 16);
  return __builtin_nontemporal_load(p);
}

// ---------------------------------------------------------------- weight prep
// WiT[g][k] = Wi[k][g] (f16), WhT[g][k] = Wh[k][g], WdT[f][h] = Wd[h][f]
__global__ void convert_weights(const float* __restrict__ Wi,
                                const float* __restrict__ Wh,
                                const float* __restrict__ Wd,
                                _Float16* __restrict__ WiT,
                                _Float16* __restrict__ WhT,
                                _Float16* __restrict__ WdT) {
  const long nWi = (long)G_DIM * F_DIM;           // 2,097,152
  const long nWh = (long)G_DIM * H_DIM;           // 4,194,304
  const long nWd = (long)F_DIM * H_DIM;           //   524,288
  const long total = nWi + nWh + nWd;
  for (long i = (long)blockIdx.x * blockDim.x + threadIdx.x; i < total;
       i += (long)gridDim.x * blockDim.x) {
    if (i < nWi) {
      long g = i / F_DIM, k = i - g * F_DIM;
      WiT[i] = (_Float16)Wi[k * G_DIM + g];
    } else if (i < nWi + nWh) {
      long j = i - nWi;
      long g = j / H_DIM, k = j - g * H_DIM;
      WhT[j] = (_Float16)Wh[k * G_DIM + g];
    } else {
      long j = i - nWi - nWh;
      long f = j / H_DIM, h = j - f * H_DIM;
      WdT[j] = (_Float16)Wd[h * F_DIM + f];
    }
  }
}

// ---------------------------------------------------------------- fused LSTM
// 16 persistent WGs, each owns 16 sequences. 8 waves; wave w owns hidden
// columns [w*128, w*128+128). Per step, per 16-col hidden tile: 4 gate
// accumulators over K = 512 (x@Wi) + 1024 (h@Wh), then elementwise update.
// h: LDS f16 double-buffered; c: LDS f32. B fragments are register
// double-buffered so global loads stay >1 WMMA-group deep in flight.
__global__ __launch_bounds__(256, 1)
void lstm_kernel(const float* __restrict__ series,
                 const _Float16* __restrict__ WiT,
                 const _Float16* __restrict__ WhT,
                 const float* __restrict__ bias,
                 float* __restrict__ hidden_out,   // [N][T][H] f32
                 float* __restrict__ last_out) {   // [N][H]    f32
  __shared__ _Float16 xbuf[16][FPAD];
  __shared__ _Float16 hbuf[2][16][HPAD];
  __shared__ float    cbuf[16][CPAD];

  const int tid   = threadIdx.x;
  const int wave  = tid >> 5;
  const int lane  = tid & 31;
  const int m16   = lane & 15;
  const int khalf = lane >> 4;
  const int r0    = blockIdx.x * 16;          // first sequence of this WG
  const int hc0   = wave * 128;               // this wave's hidden col base

  // h(t=0) = 0, c(t=0) = 0
  for (int i = tid; i < 2 * 16 * HPAD; i += blockDim.x)
    ((_Float16*)hbuf)[i] = (_Float16)0.0f;
  for (int i = tid; i < 16 * CPAD; i += blockDim.x)
    ((float*)cbuf)[i] = 0.0f;

  for (int t = 0; t < T_DIM; ++t) {
    // stage x_t (16 x 512) into LDS as f16
    for (int i = tid; i < 16 * F_DIM; i += blockDim.x) {
      int r = i >> 9, f = i & (F_DIM - 1);
      xbuf[r][f] =
          (_Float16)series[(size_t)(r0 + r) * (T_DIM * F_DIM) + (size_t)t * F_DIM + f];
    }
    __syncthreads();

    const _Float16* hcur = &hbuf[t & 1][0][0];
    _Float16* hnext      = &hbuf[(t + 1) & 1][0][0];

    for (int ht = 0; ht < 8; ++ht) {
      const int col = hc0 + ht * 16 + m16;    // hidden column of this lane
      const _Float16* bi = WiT + (size_t)(0 * H_DIM + col) * F_DIM;
      const _Float16* bf = WiT + (size_t)(1 * H_DIM + col) * F_DIM;
      const _Float16* bg = WiT + (size_t)(2 * H_DIM + col) * F_DIM;
      const _Float16* bo = WiT + (size_t)(3 * H_DIM + col) * F_DIM;
      const _Float16* ci = WhT + (size_t)(0 * H_DIM + col) * H_DIM;
      const _Float16* cf = WhT + (size_t)(1 * H_DIM + col) * H_DIM;
      const _Float16* cg = WhT + (size_t)(2 * H_DIM + col) * H_DIM;
      const _Float16* co = WhT + (size_t)(3 * H_DIM + col) * H_DIM;

      v8f acc[4];
#pragma unroll
      for (int q = 0; q < 4; ++q) {
        float bv = bias[q * H_DIM + col];
#pragma unroll
        for (int j = 0; j < 8; ++j) acc[q][j] = bv;
      }

      // ---- x_t @ Wi  (K = 512), B double-buffered in registers
      v16h B0 = load_b_nt(bi, khalf, 0);
      v16h B1 = load_b_nt(bf, khalf, 0);
      v16h B2 = load_b_nt(bg, khalf, 0);
      v16h B3 = load_b_nt(bo, khalf, 0);
      for (int k = 0; k < F_DIM; k += 32) {
        const int kn = (k + 32 < F_DIM) ? k + 32 : k;   // clamp (last redundant)
        v16h a  = load_a(&xbuf[0][0], FPAD, m16, khalf, k);
        v16h N0 = load_b_nt(bi, khalf, kn);
        v16h N1 = load_b_nt(bf, khalf, kn);
        v16h N2 = load_b_nt(bg, khalf, kn);
        v16h N3 = load_b_nt(bo, khalf, kn);
        acc[0] = wmma16(a, B0, acc[0]);
        acc[1] = wmma16(a, B1, acc[1]);
        acc[2] = wmma16(a, B2, acc[2]);
        acc[3] = wmma16(a, B3, acc[3]);
        B0 = N0; B1 = N1; B2 = N2; B3 = N3;
      }

      // ---- h @ Wh  (K = 1024), same pipeline
      B0 = load_b_nt(ci, khalf, 0);
      B1 = load_b_nt(cf, khalf, 0);
      B2 = load_b_nt(cg, khalf, 0);
      B3 = load_b_nt(co, khalf, 0);
      for (int k = 0; k < H_DIM; k += 32) {
        const int kn = (k + 32 < H_DIM) ? k + 32 : k;
        v16h a  = load_a(hcur, HPAD, m16, khalf, k);
        v16h N0 = load_b_nt(ci, khalf, kn);
        v16h N1 = load_b_nt(cf, khalf, kn);
        v16h N2 = load_b_nt(cg, khalf, kn);
        v16h N3 = load_b_nt(co, khalf, kn);
        acc[0] = wmma16(a, B0, acc[0]);
        acc[1] = wmma16(a, B1, acc[1]);
        acc[2] = wmma16(a, B2, acc[2]);
        acc[3] = wmma16(a, B3, acc[3]);
        B0 = N0; B1 = N1; B2 = N2; B3 = N3;
      }

      // gate nonlinearities + state update; C/D layout: row = j + 8*khalf
#pragma unroll
      for (int j = 0; j < 8; ++j) {
        const int mrow = khalf * 8 + j;
        float iv = fast_sigmoid(acc[0][j]);
        float fv = fast_sigmoid(acc[1][j]);
        float gv = fast_tanh(acc[2][j]);
        float ov = fast_sigmoid(acc[3][j]);
        float cv = fv * cbuf[mrow][col] + iv * gv;
        cbuf[mrow][col] = cv;
        float hv = ov * fast_tanh(cv);
        hnext[mrow * HPAD + col] = (_Float16)hv;
        hidden_out[(size_t)(r0 + mrow) * (T_DIM * H_DIM) + (size_t)t * H_DIM + col] = hv;
        if (t == T_DIM - 1)
          last_out[(size_t)(r0 + mrow) * H_DIM + col] = hv;
      }
    }
    __syncthreads();
  }
}

// ---------------------------------------------------------------- down proj
// out_seq[r][f] = hidden[r][:] @ W_down[:, f], rows = N*T = 32768, K = 1024.
// One 16x16 tile per wave; A converted f32->f16 on the fly (both streams NT:
// each byte is read exactly once per wave).
__global__ __launch_bounds__(256)
void down_proj(const float* __restrict__ hidden,
               const _Float16* __restrict__ WdT,
               float* __restrict__ out_seq) {
  const int wave  = threadIdx.x >> 5;
  const int lane  = threadIdx.x & 31;
  const int m16   = lane & 15;
  const int khalf = lane >> 4;

  const long tile = (long)blockIdx.x * 8 + wave;     // 65536 tiles
  const int  rt   = (int)(tile >> 5);                // / (512/16)
  const int  ct   = (int)(tile & 31);
  const int  r0   = rt * 16;
  const int  col  = ct * 16 + m16;

  const float*    arow = hidden + (size_t)(r0 + m16) * H_DIM;
  const _Float16* bcol = WdT + (size_t)col * H_DIM;

  v8f acc;
#pragma unroll
  for (int j = 0; j < 8; ++j) acc[j] = 0.0f;

  v16h B = load_b_nt(bcol, khalf, 0);
#pragma unroll 4
  for (int k = 0; k < H_DIM; k += 32) {
    const int kn = (k + 32 < H_DIM) ? k + 32 : k;
    const float* p = arow + k + khalf * 16;
    v8f f0 = __builtin_nontemporal_load((const v8f*)(p));
    v8f f1 = __builtin_nontemporal_load((const v8f*)(p + 8));
    v16h a;
#pragma unroll
    for (int j = 0; j < 8; ++j) {
      a[j]     = (_Float16)f0[j];
      a[j + 8] = (_Float16)f1[j];
    }
    v16h N = load_b_nt(bcol, khalf, kn);
    acc = wmma16(a, B, acc);
    B = N;
  }

#pragma unroll
  for (int j = 0; j < 8; ++j)
    out_seq[(size_t)(r0 + khalf * 8 + j) * F_DIM + col] = acc[j];
}

// ---------------------------------------------------------------- launch
extern "C" void kernel_launch(void* const* d_in, const int* in_sizes, int n_in,
                              void* d_out, int out_size, void* d_ws, size_t ws_size,
                              hipStream_t stream) {
  (void)in_sizes; (void)n_in; (void)out_size; (void)ws_size;
  const float* series = (const float*)d_in[0];
  const float* Wi     = (const float*)d_in[1];
  const float* Wh     = (const float*)d_in[2];
  const float* b      = (const float*)d_in[3];
  const float* Wd     = (const float*)d_in[4];

  float* out      = (float*)d_out;
  float* out_seq  = out;                                          // 16,777,216
  float* hidden   = out + (size_t)N_SEQ * T_DIM * F_DIM;          // 33,554,432
  float* last     = hidden + (size_t)N_SEQ * T_DIM * H_DIM;       //    262,144

  _Float16* WiT = (_Float16*)d_ws;                 //  4 MB
  _Float16* WhT = WiT + (size_t)G_DIM * F_DIM;     //  8 MB
  _Float16* WdT = WhT + (size_t)G_DIM * H_DIM;     //  1 MB

  convert_weights<<<dim3(4096), dim3(256), 0, stream>>>(Wi, Wh, Wd, WiT, WhT, WdT);
  lstm_kernel<<<dim3(N_SEQ / 16), dim3(256), 0, stream>>>(series, WiT, WhT, b,
                                                          hidden, last);
  down_proj<<<dim3((N_SEQ * T_DIM / 16) * (F_DIM / 16) / 8), dim3(256), 0, stream>>>(
      hidden, WdT, out_seq);
}
